// MultiTaskViT_82205674045979
// MI455X (gfx1250) — matrix operand order repaired and checked by reading, not compiled, not run on hardware
//
#include <hip/hip_runtime.h>
#include <hip/hip_bf16.h>
#include <math.h>

// ---------------- problem constants ----------------
#define VB 32
#define VN 196
#define VD 768
#define VH 12
#define VDH 64
#define VDEPTH 12
#define VMT (VB * VN)        // 6272 tokens total
#define VMLP (4 * VD)        // 3072
#define NMASK 98
#define LN_EPS 1e-6f

typedef __bf16 bf16_t;
typedef bf16_t bf16x16 __attribute__((ext_vector_type(16)));
typedef bf16_t bf16x8  __attribute__((ext_vector_type(8)));
typedef float  floatx8 __attribute__((ext_vector_type(8)));
typedef float  floatx4 __attribute__((ext_vector_type(4)));

// =====================================================================
// Generic batched GEMM:  C = act(alpha * A@B + bias) (+ Res)
// A: fp32 [M,K] row-major (lda), B: fp32 [K,N] (ldb) or transposed [N,K]
// Block tile 128x64, 8 waves (4Mx2N), each wave computes 32x32 via
// 2 A-fragments x 2 B-fragments -> 4 v_wmma_f32_16x16x32_bf16 per k-step
// (each fragment reused twice; 2 b128 LDS loads per WMMA).
// LDS: A row-major [128][56], B transposed [n][k] [64][56]; stride 112B is
// 16B-aligned (b128 loads) and spreads banks. Interior tiles stage with
// float4 global loads + b128 LDS stores; edge tiles use a scalar path.
// =====================================================================
#define GT_M 128
#define GT_N 64
#define GT_K 32
#define LSTR 56   // bf16 elements per LDS row (112 bytes)

__global__ __launch_bounds__(256) void vk_gemm_wmma(
    const float* __restrict__ A, long lda, long saO, long saI,
    const float* __restrict__ Bm, long ldb, long sbO, long sbI, int transB,
    const float* __restrict__ bias, const float* __restrict__ Res,
    float* __restrict__ C, long ldc, long scO, long scI,
    int M, int N, int K, float alpha, int act, int batchInner)
{
  __shared__ __align__(16) bf16_t As[GT_M][LSTR];   // [m][k]
  __shared__ __align__(16) bf16_t Bs[GT_N][LSTR];   // [n][k]  (transposed tile)

  const int z  = blockIdx.z;
  const int zo = z / batchInner;
  const int zi = z - zo * batchInner;
  A  += zo * saO + zi * saI;
  Bm += zo * sbO + zi * sbI;
  C  += zo * scO + zi * scI;
  if (Res) Res += zo * scO + zi * scI;

  const int m0   = blockIdx.y * GT_M;
  const int n0   = blockIdx.x * GT_N;
  const int tid  = threadIdx.x;
  const int lane = tid & 31;
  const int wave = tid >> 5;
  const int wm   = wave & 3;   // 0..3 -> 32-row slabs
  const int wn   = wave >> 2;  // 0..1 -> 32-col slabs

  // staging coordinates
  const int am = tid >> 1;            // 0..127 (A row)
  const int ak = (tid & 1) * 16;      // 0,16
  const int bn = tid & 63;            // 0..63  (B col -> LDS row)
  const int bk = (tid >> 6) * 8;      // 0,8,16,24

  const bool aRows = (m0 + GT_M <= M);
  const bool bCols = (n0 + GT_N <= N);

  floatx8 acc00 = {}, acc01 = {}, acc10 = {}, acc11 = {};

  const int kTiles = (K + GT_K - 1) / GT_K;
  for (int kt = 0; kt < kTiles; ++kt) {
    const int k0 = kt * GT_K;
    const bool kFull = (k0 + GT_K <= K);

    // ---- stage A tile [128m x 32k], 16 elems/thread ----
    if (aRows && kFull) {
      const float* ap = A + (long)(m0 + am) * lda + (k0 + ak);
      floatx4 f0 = *(const floatx4*)ap;
      floatx4 f1 = *(const floatx4*)(ap + 4);
      floatx4 f2 = *(const floatx4*)(ap + 8);
      floatx4 f3 = *(const floatx4*)(ap + 12);
      bf16x8 v0, v1;
      v0[0] = (bf16_t)f0[0]; v0[1] = (bf16_t)f0[1];
      v0[2] = (bf16_t)f0[2]; v0[3] = (bf16_t)f0[3];
      v0[4] = (bf16_t)f1[0]; v0[5] = (bf16_t)f1[1];
      v0[6] = (bf16_t)f1[2]; v0[7] = (bf16_t)f1[3];
      v1[0] = (bf16_t)f2[0]; v1[1] = (bf16_t)f2[1];
      v1[2] = (bf16_t)f2[2]; v1[3] = (bf16_t)f2[3];
      v1[4] = (bf16_t)f3[0]; v1[5] = (bf16_t)f3[1];
      v1[6] = (bf16_t)f3[2]; v1[7] = (bf16_t)f3[3];
      *(bf16x8*)&As[am][ak]     = v0;
      *(bf16x8*)&As[am][ak + 8] = v1;
    } else {
      bf16x8 v0, v1;
#pragma unroll
      for (int j = 0; j < 8; ++j) {
        int gm = m0 + am;
        int gk0 = k0 + ak + j, gk1 = k0 + ak + 8 + j;
        float f0 = (gm < M && gk0 < K) ? A[(long)gm * lda + gk0] : 0.0f;
        float f1 = (gm < M && gk1 < K) ? A[(long)gm * lda + gk1] : 0.0f;
        v0[j] = (bf16_t)f0;
        v1[j] = (bf16_t)f1;
      }
      *(bf16x8*)&As[am][ak]     = v0;
      *(bf16x8*)&As[am][ak + 8] = v1;
    }

    // ---- stage B tile, transposed into LDS [64n x 32k] ----
    if (bCols && kFull) {
      bf16x8 v;
      if (transB) {
        const float* bp = Bm + (long)(n0 + bn) * ldb + (k0 + bk);
        floatx4 f0 = *(const floatx4*)bp;
        floatx4 f1 = *(const floatx4*)(bp + 4);
        v[0] = (bf16_t)f0[0]; v[1] = (bf16_t)f0[1];
        v[2] = (bf16_t)f0[2]; v[3] = (bf16_t)f0[3];
        v[4] = (bf16_t)f1[0]; v[5] = (bf16_t)f1[1];
        v[6] = (bf16_t)f1[2]; v[7] = (bf16_t)f1[3];
      } else {
#pragma unroll
        for (int j = 0; j < 8; ++j)
          v[j] = (bf16_t)Bm[(long)(k0 + bk + j) * ldb + (n0 + bn)];
      }
      *(bf16x8*)&Bs[bn][bk] = v;
    } else {
      bf16x8 v;
#pragma unroll
      for (int j = 0; j < 8; ++j) {
        int gk = k0 + bk + j, gn = n0 + bn;
        float f = 0.0f;
        if (gk < K && gn < N)
          f = transB ? Bm[(long)gn * ldb + gk] : Bm[(long)gk * ldb + gn];
        v[j] = (bf16_t)f;
      }
      *(bf16x8*)&Bs[bn][bk] = v;
    }
    __syncthreads();

    // ---- fragments: b128 pairs at k = 8*hi and 16 + 8*hi ----
    const int hi  = lane >> 4;
    const int ml0 = wm * 32 + (lane & 15);
    const int ml1 = ml0 + 16;
    const int nl  = wn * 32 + (lane & 15);

    bf16x8 a00 = *(const bf16x8*)&As[ml0][hi * 8];
    bf16x8 a01 = *(const bf16x8*)&As[ml0][16 + hi * 8];
    bf16x16 af0 = __builtin_shufflevector(a00, a01, 0, 1, 2, 3, 4, 5, 6, 7,
                                          8, 9, 10, 11, 12, 13, 14, 15);
    bf16x8 a10 = *(const bf16x8*)&As[ml1][hi * 8];
    bf16x8 a11 = *(const bf16x8*)&As[ml1][16 + hi * 8];
    bf16x16 af1 = __builtin_shufflevector(a10, a11, 0, 1, 2, 3, 4, 5, 6, 7,
                                          8, 9, 10, 11, 12, 13, 14, 15);

    bf16x8 b00 = *(const bf16x8*)&Bs[nl][hi * 8];
    bf16x8 b01 = *(const bf16x8*)&Bs[nl][16 + hi * 8];
    bf16x16 bf0 = __builtin_shufflevector(b00, b01, 0, 1, 2, 3, 4, 5, 6, 7,
                                          8, 9, 10, 11, 12, 13, 14, 15);
    bf16x8 b10 = *(const bf16x8*)&Bs[nl + 16][hi * 8];
    bf16x8 b11 = *(const bf16x8*)&Bs[nl + 16][16 + hi * 8];
    bf16x16 bf1 = __builtin_shufflevector(b10, b11, 0, 1, 2, 3, 4, 5, 6, 7,
                                          8, 9, 10, 11, 12, 13, 14, 15);

    acc00 = __builtin_amdgcn_wmma_f32_16x16x32_bf16(false, af0, false, bf0,
                                                    (short)0, acc00, false, false);
    acc01 = __builtin_amdgcn_wmma_f32_16x16x32_bf16(false, af0, false, bf1,
                                                    (short)0, acc01, false, false);
    acc10 = __builtin_amdgcn_wmma_f32_16x16x32_bf16(false, af1, false, bf0,
                                                    (short)0, acc10, false, false);
    acc11 = __builtin_amdgcn_wmma_f32_16x16x32_bf16(false, af1, false, bf1,
                                                    (short)0, acc11, false, false);
    __syncthreads();
  }

  // epilogue: f32 16x16 C/D layout (vgpr r -> row (lane>>4)*8+r, col lane&15)
  const int hi  = lane >> 4;
  const int nlo = lane & 15;
#pragma unroll
  for (int sub = 0; sub < 2; ++sub) {
#pragma unroll
    for (int r = 0; r < 8; ++r) {
      int mc = m0 + wm * 32 + sub * 16 + hi * 8 + r;
      if (mc >= M) continue;
#pragma unroll
      for (int t = 0; t < 2; ++t) {
        int nc = n0 + wn * 32 + t * 16 + nlo;
        if (nc >= N) continue;
        float val;
        if (sub == 0) val = (t ? acc01[r] : acc00[r]);
        else          val = (t ? acc11[r] : acc10[r]);
        val *= alpha;
        if (bias) val += bias[nc];
        if (act == 1) val = 0.5f * val * (1.0f + erff(val * 0.70710678118654752f));
        if (Res) val += Res[(long)mc * ldc + nc];
        C[(long)mc * ldc + nc] = val;
      }
    }
  }
}

// ---------------- LayerNorm (one block per 768-wide row) ----------------
__global__ __launch_bounds__(256) void vk_layernorm(
    const float* __restrict__ x, const float* __restrict__ g,
    const float* __restrict__ b, float* __restrict__ y, int cols)
{
  const long row = blockIdx.x;
  const float* xr = x + row * cols;
  float s = 0.f, sq = 0.f;
  for (int c = threadIdx.x; c < cols; c += 256) {
    float v = xr[c]; s += v; sq += v * v;
  }
  for (int o = 16; o; o >>= 1) { s += __shfl_xor(s, o); sq += __shfl_xor(sq, o); }
  __shared__ float rs[8], rq[8];
  const int lane = threadIdx.x & 31, wave = threadIdx.x >> 5;
  if (lane == 0) { rs[wave] = s; rq[wave] = sq; }
  __syncthreads();
  if (wave == 0) {
    s  = (lane < 8) ? rs[lane] : 0.f;
    sq = (lane < 8) ? rq[lane] : 0.f;
    for (int o = 4; o; o >>= 1) { s += __shfl_xor(s, o); sq += __shfl_xor(sq, o); }
    if (lane == 0) { rs[0] = s; rq[0] = sq; }
  }
  __syncthreads();
  const float mean = rs[0] / cols;
  const float var  = rq[0] / cols - mean * mean;
  const float inv  = rsqrtf(var + LN_EPS);
  float* yr = y + row * cols;
  for (int c = threadIdx.x; c < cols; c += 256)
    yr[c] = (xr[c] - mean) * inv * g[c] + b[c];
}

// ---------------- softmax: one wave32 per row ----------------
__global__ __launch_bounds__(256) void vk_softmax_rows(
    float* __restrict__ s, int rows, int cols)
{
  const int row  = blockIdx.x * 8 + (threadIdx.x >> 5);
  const int lane = threadIdx.x & 31;
  if (row >= rows) return;
  float* p = s + (long)row * cols;
  float mx = -3.0e38f;
  for (int c = lane; c < cols; c += 32) mx = fmaxf(mx, p[c]);
  for (int o = 16; o; o >>= 1) mx = fmaxf(mx, __shfl_xor(mx, o));
  float sum = 0.f;
  for (int c = lane; c < cols; c += 32) { float e = expf(p[c] - mx); p[c] = e; sum += e; }
  for (int o = 16; o; o >>= 1) sum += __shfl_xor(sum, o);
  const float inv = 1.0f / sum;
  for (int c = lane; c < cols; c += 32) p[c] *= inv;
}

// ---------------- patchify: [B,C,224,224] -> [B*N, 768] ----------------
__global__ __launch_bounds__(256) void vk_patchify(
    const float* __restrict__ img, float* __restrict__ X)
{
  const long total = (long)VMT * VD;
  for (long idx = (long)blockIdx.x * 256 + threadIdx.x; idx < total;
       idx += (long)gridDim.x * 256) {
    int col = (int)(idx % VD);
    long row = idx / VD;
    int b = (int)(row / VN), n = (int)(row % VN);
    int gy = n / 14, gx = n % 14;
    int c = col >> 8, r = col & 255;
    int ph = r >> 4, pw = r & 15;
    X[idx] = img[(((long)b * 3 + c) * 224 + gy * 16 + ph) * 224 + gx * 16 + pw];
  }
}

__global__ __launch_bounds__(256) void vk_add_pos(
    float* __restrict__ z, const float* __restrict__ pos)
{
  const long total = (long)VMT * VD;
  for (long idx = (long)blockIdx.x * 256 + threadIdx.x; idx < total;
       idx += (long)gridDim.x * 256) {
    long row = idx / VD;
    int d = (int)(idx % VD);
    z[idx] += pos[(row % VN) * VD + d];
  }
}

// mean over tokens: out[b,d] = mean_n feats[b,n,d]
__global__ __launch_bounds__(256) void vk_mean_tokens(
    const float* __restrict__ f, float* __restrict__ out)
{
  for (int idx = blockIdx.x * 256 + threadIdx.x; idx < VB * VD;
       idx += gridDim.x * 256) {
    int b = idx / VD, d = idx % VD;
    float s = 0.f;
    for (int n = 0; n < VN; ++n) s += f[((long)b * VN + n) * VD + d];
    out[idx] = s * (1.0f / VN);
  }
}

// avg over visible (unmasked) tokens
__global__ __launch_bounds__(256) void vk_avg_vis(
    const float* __restrict__ f, const unsigned char* __restrict__ mask,
    float* __restrict__ out)
{
  for (int idx = blockIdx.x * 256 + threadIdx.x; idx < VB * VD;
       idx += gridDim.x * 256) {
    int b = idx / VD, d = idx % VD;
    float s = 0.f;
    for (int n = 0; n < VN; ++n)
      if (!mask[b * VN + n]) s += f[((long)b * VN + n) * VD + d];
    out[idx] = s * (1.0f / (VN - NMASK));
  }
}

// small dense C[M,N] = A[M,K] @ W[K,N] + bias (tiny head GEMMs)
__global__ __launch_bounds__(256) void vk_small_matmul(
    const float* __restrict__ A, const float* __restrict__ W,
    const float* __restrict__ bias, float* __restrict__ C, int M, int N, int K)
{
  int idx = blockIdx.x * 256 + threadIdx.x;
  if (idx >= M * N) return;
  int m = idx / N, n = idx % N;
  float s = bias ? bias[n] : 0.f;
  for (int k = 0; k < K; ++k) s += A[(long)m * K + k] * W[(long)k * N + n];
  C[idx] = s;
}

// masked-token indices, ascending (exactly NMASK true per row)
__global__ void vk_midx(const unsigned char* __restrict__ mask, int* __restrict__ midx)
{
  int b = blockIdx.x * blockDim.x + threadIdx.x;
  if (b >= VB) return;
  int c = 0;
  for (int n = 0; n < VN; ++n)
    if (mask[b * VN + n] && c < NMASK) midx[b * NMASK + c++] = n;
}

// rppp_in[b] = concat(feat[b, a_idx[b]], feat[b, b_idx[b]])
__global__ __launch_bounds__(256) void vk_gather_rppp(
    const float* __restrict__ f, const int* __restrict__ ai,
    const int* __restrict__ bi, float* __restrict__ out)
{
  for (int idx = blockIdx.x * 256 + threadIdx.x; idx < VB * 2 * VD;
       idx += gridDim.x * 256) {
    int b = idx / (2 * VD), d2 = idx % (2 * VD);
    int row = (d2 < VD) ? ai[b] : bi[b];
    int d = d2 % VD;
    out[idx] = f[((long)b * VN + row) * VD + d];
  }
}

// out4[b,i,d] = tgt_feats[b, midx[b,i], d]
__global__ __launch_bounds__(256) void vk_gather_masked(
    const float* __restrict__ f, const int* __restrict__ midx,
    float* __restrict__ out)
{
  const long total = (long)VB * NMASK * VD;
  for (long idx = (long)blockIdx.x * 256 + threadIdx.x; idx < total;
       idx += (long)gridDim.x * 256) {
    int d = (int)(idx % VD);
    long bi = idx / VD;
    int b = (int)(bi / NMASK), i = (int)(bi % NMASK);
    out[idx] = f[((long)b * VN + midx[b * NMASK + i]) * VD + d];
  }
}

// out3[b,i,d] = pred[b,d]
__global__ __launch_bounds__(256) void vk_bcast_pred(
    const float* __restrict__ pred, float* __restrict__ out)
{
  const long total = (long)VB * NMASK * VD;
  for (long idx = (long)blockIdx.x * 256 + threadIdx.x; idx < total;
       idx += (long)gridDim.x * 256) {
    int d = (int)(idx % VD);
    int b = (int)(idx / ((long)NMASK * VD));
    out[idx] = pred[b * VD + d];
  }
}

// =====================================================================
// Host-side orchestration
// =====================================================================
struct VitParams {
  const float *patch_w, *patch_b, *pos, *ln_g, *ln_b;
  const float *ln1_g[VDEPTH], *ln1_b[VDEPTH], *qkv_w[VDEPTH], *qkv_b[VDEPTH],
              *proj_w[VDEPTH], *proj_b[VDEPTH], *ln2_g[VDEPTH], *ln2_b[VDEPTH],
              *fc1_w[VDEPTH], *fc1_b[VDEPTH], *fc2_w[VDEPTH], *fc2_b[VDEPTH];
};

static VitParams load_params(void* const* d_in, int base) {
  VitParams P;
  int i = base;
  P.patch_w = (const float*)d_in[i++];
  P.patch_b = (const float*)d_in[i++];
  P.pos     = (const float*)d_in[i++];
  P.ln_g    = (const float*)d_in[i++];
  P.ln_b    = (const float*)d_in[i++];
  for (int b = 0; b < VDEPTH; ++b) {
    P.ln1_g[b]  = (const float*)d_in[i++];
    P.ln1_b[b]  = (const float*)d_in[i++];
    P.qkv_w[b]  = (const float*)d_in[i++];
    P.qkv_b[b]  = (const float*)d_in[i++];
    P.proj_w[b] = (const float*)d_in[i++];
    P.proj_b[b] = (const float*)d_in[i++];
    P.ln2_g[b]  = (const float*)d_in[i++];
    P.ln2_b[b]  = (const float*)d_in[i++];
    P.fc1_w[b]  = (const float*)d_in[i++];
    P.fc1_b[b]  = (const float*)d_in[i++];
    P.fc2_w[b]  = (const float*)d_in[i++];
    P.fc2_b[b]  = (const float*)d_in[i++];
  }
  return P;
}

static void launch_gemm(hipStream_t st,
                        const float* A, long lda, long saO, long saI,
                        const float* Bm, long ldb, long sbO, long sbI, int transB,
                        const float* bias, const float* Res,
                        float* C, long ldc, long scO, long scI,
                        int M, int N, int K, float alpha, int act,
                        int zOuter, int zInner)
{
  dim3 grid((N + GT_N - 1) / GT_N, (M + GT_M - 1) / GT_M, zOuter * zInner);
  vk_gemm_wmma<<<grid, 256, 0, st>>>(A, lda, saO, saI, Bm, ldb, sbO, sbI, transB,
                                     bias, Res, C, ldc, scO, scI,
                                     M, N, K, alpha, act, zInner);
}

static void run_vit(const VitParams& P, const float* img,
                    float* z, float* h, float* qkv, float* s, float* o, float* t,
                    float* feats, hipStream_t st)
{
  const long QKS = (long)VN * 3 * VD;  // qkv token stride within one image
  vk_patchify<<<dim3((VMT * VD + 255) / 256), 256, 0, st>>>(img, h);
  launch_gemm(st, h, VD, 0, 0, P.patch_w, VD, 0, 0, 0, P.patch_b, nullptr,
              z, VD, 0, 0, VMT, VD, VD, 1.0f, 0, 1, 1);
  vk_add_pos<<<dim3((VMT * VD + 255) / 256), 256, 0, st>>>(z, P.pos);

  for (int b = 0; b < VDEPTH; ++b) {
    vk_layernorm<<<dim3(VMT), 256, 0, st>>>(z, P.ln1_g[b], P.ln1_b[b], h, VD);
    launch_gemm(st, h, VD, 0, 0, P.qkv_w[b], 3 * VD, 0, 0, 0, P.qkv_b[b], nullptr,
                qkv, 3 * VD, 0, 0, VMT, 3 * VD, VD, 1.0f, 0, 1, 1);
    // S = Q K^T / sqrt(dh), batched over (b,h)
    launch_gemm(st, qkv, QKS, (long)VN * QKS, VDH,
                qkv + VD, QKS, (long)VN * QKS, VDH, 1,
                nullptr, nullptr,
                s, VN, (long)VH * VN * VN, (long)VN * VN,
                VN, VN, VDH, 0.125f, 0, VB, VH);
    vk_softmax_rows<<<dim3((VB * VH * VN + 7) / 8), 256, 0, st>>>(s, VB * VH * VN, VN);
    // O = P V, batched over (b,h), packed back into [B*N, D]
    launch_gemm(st, s, VN, (long)VH * VN * VN, (long)VN * VN,
                qkv + 2 * VD, QKS, (long)VN * QKS, VDH, 0,
                nullptr, nullptr,
                o, VD, (long)VN * VD, VDH,
                VN, VDH, VN, 1.0f, 0, VB, VH);
    launch_gemm(st, o, VD, 0, 0, P.proj_w[b], VD, 0, 0, 0, P.proj_b[b], z,
                z, VD, 0, 0, VMT, VD, VD, 1.0f, 0, 1, 1);
    vk_layernorm<<<dim3(VMT), 256, 0, st>>>(z, P.ln2_g[b], P.ln2_b[b], h, VD);
    launch_gemm(st, h, VD, 0, 0, P.fc1_w[b], VMLP, 0, 0, 0, P.fc1_b[b], nullptr,
                t, VMLP, 0, 0, VMT, VMLP, VD, 1.0f, 1, 1, 1);
    launch_gemm(st, t, VMLP, 0, 0, P.fc2_w[b], VD, 0, 0, 0, P.fc2_b[b], z,
                z, VD, 0, 0, VMT, VD, VMLP, 1.0f, 0, 1, 1);
  }
  vk_layernorm<<<dim3(VMT), 256, 0, st>>>(z, P.ln_g, P.ln_b, feats, VD);
}

extern "C" void kernel_launch(void* const* d_in, const int* in_sizes, int n_in,
                              void* d_out, int out_size, void* d_ws, size_t ws_size,
                              hipStream_t stream)
{
  (void)in_sizes; (void)n_in; (void)out_size; (void)ws_size;
  const float* orig_img = (const float*)d_in[0];
  const float* rot_img  = (const float*)d_in[1];
  const int*   a_idx    = (const int*)d_in[2];
  const int*   b_idx    = (const int*)d_in[3];
  const unsigned char* mask = (const unsigned char*)d_in[4];
  VitParams Penc = load_params(d_in, 5);
  VitParams Ptgt = load_params(d_in, 5 + 5 + 12 * VDEPTH);
  int hb = 5 + 2 * (5 + 12 * VDEPTH);  // 303
  const float* rot_w   = (const float*)d_in[hb + 0];
  const float* rot_b   = (const float*)d_in[hb + 1];
  const float* rppp_w1 = (const float*)d_in[hb + 2];
  const float* rppp_b1 = (const float*)d_in[hb + 3];
  const float* rppp_w2 = (const float*)d_in[hb + 4];
  const float* rppp_b2 = (const float*)d_in[hb + 5];
  const float* rec_w   = (const float*)d_in[hb + 6];
  const float* rec_b   = (const float*)d_in[hb + 7];

  float* out = (float*)d_out;
  float* out_rot  = out;                                  // [32,4]
  float* out_rppp = out + VB * 4;                          // [32,8]
  float* out_pred = out + VB * 4 + VB * 8;                 // [32,98,768]
  float* out_tgt  = out_pred + (long)VB * NMASK * VD;      // [32,98,768]

  char* ws = (char*)d_ws;
  size_t off = 0;
  auto carve = [&](size_t bytes) {
    void* p = ws + off;
    off += (bytes + 255) & ~(size_t)255;
    return p;
  };
  float* z    = (float*)carve((size_t)VMT * VD * 4);
  float* h    = (float*)carve((size_t)VMT * VD * 4);
  size_t qkvB = (size_t)VMT * 3 * VD * 4;
  size_t sB   = (size_t)VB * VH * VN * VN * 4;
  size_t tB   = (size_t)VMT * VMLP * 4;
  size_t uB   = qkvB + sB;  if (tB > uB) uB = tB;
  char*  U    = (char*)carve(uB);
  float* qkv  = (float*)U;
  float* satt = (float*)(U + qkvB);
  float* tbuf = (float*)U;
  float* o    = (float*)carve((size_t)VMT * VD * 4);
  float* feat = (float*)carve((size_t)VMT * VD * 4);
  float* mrot = (float*)carve((size_t)VB * VD * 4);
  float* rin  = (float*)carve((size_t)VB * 2 * VD * 4);
  float* rhid = (float*)carve((size_t)VB * VD * 4);
  float* avg  = (float*)carve((size_t)VB * VD * 4);
  float* pred = (float*)carve((size_t)VB * VD * 4);
  int*   midx = (int*)carve((size_t)VB * NMASK * 4);

  const long bcast = ((long)VB * NMASK * VD + 255) / 256;

  // 1) rotation head: encoder on rot_img
  run_vit(Penc, rot_img, z, h, qkv, satt, o, tbuf, /*feats=*/o, stream);
  vk_mean_tokens<<<dim3(96), 256, 0, stream>>>(o, mrot);
  vk_small_matmul<<<dim3(1), 256, 0, stream>>>(mrot, rot_w, rot_b, out_rot,
                                               VB, 4, VD);

  // 2) RPPP head + reconstruction source: encoder on orig_img
  run_vit(Penc, orig_img, z, h, qkv, satt, o, tbuf, /*feats=*/feat, stream);
  vk_gather_rppp<<<dim3(192), 256, 0, stream>>>(feat, a_idx, b_idx, rin);
  launch_gemm(stream, rin, 2 * VD, 0, 0, rppp_w1, VD, 0, 0, 0, rppp_b1, nullptr,
              rhid, VD, 0, 0, VB, VD, 2 * VD, 1.0f, /*gelu*/1, 1, 1);
  vk_small_matmul<<<dim3(1), 256, 0, stream>>>(rhid, rppp_w2, rppp_b2, out_rppp,
                                               VB, 8, VD);
  vk_avg_vis<<<dim3(96), 256, 0, stream>>>(feat, mask, avg);
  launch_gemm(stream, avg, VD, 0, 0, rec_w, VD, 0, 0, 0, rec_b, nullptr,
              pred, VD, 0, 0, VB, VD, VD, 1.0f, 0, 1, 1);
  vk_midx<<<dim3(1), 32, 0, stream>>>(mask, midx);
  vk_bcast_pred<<<dim3((unsigned)bcast), 256, 0, stream>>>(pred, out_pred);

  // 3) target encoder on orig_img, gather masked tokens
  run_vit(Ptgt, orig_img, z, h, qkv, satt, o, tbuf, /*feats=*/o, stream);
  vk_gather_masked<<<dim3((unsigned)bcast), 256, 0, stream>>>(o, midx, out_tgt);
}
